// FieldFormerAutograd_2568390443120
// MI455X (gfx1250) — compile-verified, matrix operand order checked by reading, not run on hardware
//
#include <hip/hip_runtime.h>
#include <hip/hip_bf16.h>
#include <math.h>

typedef _Float16 half_t;
typedef __attribute__((ext_vector_type(16))) _Float16 v16h;
typedef __attribute__((ext_vector_type(8)))  _Float16 v8h;
typedef __attribute__((ext_vector_type(8)))  float    v8f;

#define NXg 128
#define NYg 128
#define NTg 64
#define KNB 128          // sequence length S
#define DM  64           // model dim
#define NHEAD 4
#define NLAYER 2
#define EPSf 1e-5f

// ---- f16 weight arena offsets (in halves) inside d_ws ----
#define W16_QKV 0                       // 2*192*64 = 24576
#define W16_OUT 24576                   // 2*64*64  = 8192
#define W16_FF1 32768                   // 2*128*64 = 16384
#define W16_FF2 49152                   // 2*64*128 = 16384
#define W16_TOT 65536                   // halves -> 128 KB

// ---- LDS arena offsets (bytes) ----
#define OFF_H    0                      // f32 [128][64]  32768
#define OFF_H16  32768                  // f16 [128][64]  16384
#define OFF_QK   49152                  // f16 [128][128] 32768 (q cols 0-63, k cols 64-127; alias tok)
#define OFF_VT   81920                  // f16 [64][128]  16384 (V transposed: [d][token])
#define OFF_O16  98304                  // f16 [128][64]  16384 (attn out)
#define OFF_B    114688                 // f32 biases: 192+64+128+64 = 448 -> 1792 B
#define OFF_G    116480                 // f32 [128] head scratch
#define SMEM_SZ  116992

// ---------- WMMA fragment helpers (ISA 7.12.2 layouts, wave32) ----------
// A (16x32 f16) and B (32x16 f16 from row-major [N][K]):
//   lane L holds row/col (L&15); halves e<8 -> K = e + ((L&16)?8:0),
//   halves e>=8 -> K = 16 + (e-8) + ((L&16)?8:0). Two 16B loads per lane.
__device__ __forceinline__ v16h frag_ld(const half_t* base, int rowbase, int stride,
                                        int koff, int lane) {
  int r = rowbase + (lane & 15);
  int c = (lane & 16) ? 8 : 0;
  const half_t* p = base + r * stride + koff + c;
  v8h lo = *(const v8h*)(p);
  v8h hi = *(const v8h*)(p + 16);
  v16h out;
#pragma unroll
  for (int e = 0; e < 8; ++e) { out[e] = lo[e]; out[e + 8] = hi[e]; }
  return out;
}

// K=16 fragment zero-padded to K=32 (per-head attention, dh=16)
__device__ __forceinline__ v16h frag_ld_k16(const half_t* base, int rowbase, int stride,
                                            int koff, int lane) {
  int r = rowbase + (lane & 15);
  int c = (lane & 16) ? 8 : 0;
  v8h lo = *(const v8h*)(base + r * stride + koff + c);
  v16h out;
#pragma unroll
  for (int e = 0; e < 8; ++e) { out[e] = lo[e]; out[e + 8] = (half_t)0.0f; }
  return out;
}

__device__ __forceinline__ v8f wmma_f16(v16h a, v16h b, v8f c) {
  return __builtin_amdgcn_wmma_f32_16x16x32_f16(false, a, false, b, (short)0, c,
                                                false, false);
}

__device__ __forceinline__ float dper(float diff, float L) {
  float m = fmodf(diff + 0.5f * L, L);
  m = (m < 0.0f) ? m + L : m;
  return m - 0.5f * L;
}

// ---------- weight f32 -> f16 conversion into d_ws ----------
__global__ void ff_convert_weights(const float* qkv_w, const float* out_w,
                                   const float* ff1_w, const float* ff2_w,
                                   half_t* w16) {
  int i = blockIdx.x * blockDim.x + threadIdx.x;
  if (i < W16_OUT)           w16[i] = (half_t)qkv_w[i];
  else if (i < W16_FF1)      w16[i] = (half_t)out_w[i - W16_OUT];
  else if (i < W16_FF2)      w16[i] = (half_t)ff1_w[i - W16_FF1];
  else if (i < W16_TOT)      w16[i] = (half_t)ff2_w[i - W16_FF2];
}

// ---------- fused per-query transformer ----------
__global__ __launch_bounds__(256)
void ff_main(const int* __restrict__ q_lin_idx, const int* __restrict__ offs,
             const float* __restrict__ coords, const float* __restrict__ vals,
             const float* __restrict__ log_gammas,
             const float* __restrict__ proj_w, const float* __restrict__ proj_b,
             const float* __restrict__ qkv_b, const float* __restrict__ out_b,
             const float* __restrict__ ln1_s, const float* __restrict__ ln1_b,
             const float* __restrict__ ff1_b, const float* __restrict__ ff2_b,
             const float* __restrict__ ln2_s, const float* __restrict__ ln2_b,
             const float* __restrict__ head_ln_s, const float* __restrict__ head_ln_b,
             const float* __restrict__ head1_w, const float* __restrict__ head1_b,
             const float* __restrict__ head2_w, const float* __restrict__ head2_b,
             const half_t* __restrict__ w16, float* __restrict__ out) {
  __shared__ __align__(128) char smem[SMEM_SZ];
  float*  hbuf = (float*)(smem + OFF_H);       // f32 residual stream [128][64]
  half_t* h16  = (half_t*)(smem + OFF_H16);    // f16 mirror of hbuf
  half_t* qk16 = (half_t*)(smem + OFF_QK);     // [128][128]: q | k
  half_t* vT16 = (half_t*)(smem + OFF_VT);     // [64][128]: V transposed
  half_t* o16  = (half_t*)(smem + OFF_O16);    // attn out [128][64]
  float*  tokb = (float*)(smem + OFF_QK);      // alias: tokens [128][6]
  float*  bqs  = (float*)(smem + OFF_B);       // qkv bias [192]
  float*  bos  = bqs + 192;                    // out bias [64]
  float*  bf1s = bos + 64;                     // ff1 bias [128]
  float*  bf2s = bf1s + 128;                   // ff2 bias [64]
  float*  gbuf = (float*)(smem + OFF_G);       // pooled vec [64]
  float*  ybuf = gbuf + 64;                    // head1 out [64]

  const int blk  = blockIdx.x;
  const int tid  = threadIdx.x;
  const int lane = tid & 31;
  const int mt   = tid >> 5;              // wave id == query M-tile
  const int loff = (lane & 16) ? 8 : 0;   // D-matrix row offset, upper half-wave
  const int ncol = lane & 15;             // D-matrix column within tile

  // ---- stage 0: neighbor gather + relative coords ----
  const int qi = q_lin_idx[blk];
  const int ii = qi / (NYg * NTg);
  const int rr = qi % (NYg * NTg);
  const int jj = rr / NTg;
  const int kk = rr % NTg;
  const float gx = __expf(log_gammas[0]);
  const float gy = __expf(log_gammas[1]);
  const float gz = __expf(log_gammas[2]);
  const float qx = coords[qi * 3 + 0], qy = coords[qi * 3 + 1], qz = coords[qi * 3 + 2];

  if (tid < KNB) {
    int di = offs[tid * 3 + 0], dj = offs[tid * 3 + 1], dk = offs[tid * 3 + 2];
    int I = (ii + di) % NXg;
    int J = (jj + dj) % NYg;
    int T = min(max(kk + dk, 0), NTg - 1);
    int nb = I * (NYg * NTg) + J * NTg + T;
    float nx = coords[nb * 3 + 0], ny = coords[nb * 3 + 1], nz = coords[nb * 3 + 2];
    tokb[tid * 6 + 0] = dper(nx - qx, 2.0f) * gx;
    tokb[tid * 6 + 1] = dper(ny - qy, 2.0f) * gy;
    tokb[tid * 6 + 2] = (nz - qz) * gz;
    tokb[tid * 6 + 3] = vals[nb * 3 + 0];
    tokb[tid * 6 + 4] = vals[nb * 3 + 1];
    tokb[tid * 6 + 5] = vals[nb * 3 + 2];
  }
  __syncthreads();

  // ---- input projection 6 -> 64 (scalar; vectorized f16 stores) ----
  for (int ch = tid; ch < KNB * 8; ch += 256) {   // 1024 chunks of 8 dims
    int r = ch >> 3, db = (ch & 7) * 8;
    v8h hv;
#pragma unroll
    for (int u = 0; u < 8; ++u) {
      int d = db + u;
      float acc = proj_b[d];
#pragma unroll
      for (int c = 0; c < 6; ++c) acc += tokb[r * 6 + c] * proj_w[d * 6 + c];
      hbuf[r * DM + d] = acc;
      hv[u] = (half_t)acc;
    }
    *(v8h*)(h16 + r * DM + db) = hv;
  }
  __syncthreads();

  const half_t* wqkv = w16 + W16_QKV;
  const half_t* wout = w16 + W16_OUT;
  const half_t* wff1 = w16 + W16_FF1;
  const half_t* wff2 = w16 + W16_FF2;

  for (int l = 0; l < NLAYER; ++l) {
    // ---- preload layer biases to LDS ----
    for (int i = tid; i < 192; i += 256) bqs[i] = qkv_b[l * 192 + i];
    if (tid < 64)  bos[tid]  = out_b[l * 64 + tid];
    if (tid < 128) bf1s[tid] = ff1_b[l * 128 + tid];
    if (tid < 64)  bf2s[tid] = ff2_b[l * 64 + tid];
    __syncthreads();

    // ---- QKV: [128x64] @ [64x192]; V stored transposed with vector stores ----
    {
      const half_t* Wq = wqkv + l * 192 * 64;
      v16h a0 = frag_ld(h16, mt * 16, DM, 0, lane);
      v16h a1 = frag_ld(h16, mt * 16, DM, 32, lane);
      for (int nt = 0; nt < 8; ++nt) {       // q,k tiles -> row-major qk16
        v8f acc = {0.f, 0.f, 0.f, 0.f, 0.f, 0.f, 0.f, 0.f};
        v16h b0 = frag_ld(Wq, nt * 16, DM, 0, lane);
        v16h b1 = frag_ld(Wq, nt * 16, DM, 32, lane);
        acc = wmma_f16(a0, b0, acc);
        acc = wmma_f16(a1, b1, acc);
        int col = nt * 16 + ncol;
        float bb = bqs[col];
#pragma unroll
        for (int v = 0; v < 8; ++v)
          qk16[(mt * 16 + loff + v) * 128 + col] = (half_t)(acc[v] + bb);
      }
      for (int nt = 8; nt < 12; ++nt) {      // v tiles -> transposed, b128 stores
        v8f acc = {0.f, 0.f, 0.f, 0.f, 0.f, 0.f, 0.f, 0.f};
        v16h b0 = frag_ld(Wq, nt * 16, DM, 0, lane);
        v16h b1 = frag_ld(Wq, nt * 16, DM, 32, lane);
        acc = wmma_f16(a0, b0, acc);
        acc = wmma_f16(a1, b1, acc);
        int col = nt * 16 + ncol;            // 128..191
        float bb = bqs[col];
        v8h pk;
#pragma unroll
        for (int v = 0; v < 8; ++v) pk[v] = (half_t)(acc[v] + bb);
        *(v8h*)(vT16 + (col - 128) * 128 + mt * 16 + loff) = pk;
      }
    }
    __syncthreads();

    // ---- attention: S^T tiles so softmax + P@V A-frags stay in registers ----
    const half_t* q16p = qk16;        // [m][d], cols 0..63
    const half_t* k16p = qk16 + 64;   // [n][d], cols 64..127
    const float scale = 0.25f;        // 1/sqrt(16)
    for (int hh = 0; hh < NHEAD; ++hh) {
      // B = this wave's Q rows (lane = query col of S^T)
      v16h bq = frag_ld_k16(q16p, mt * 16, 128, hh * 16, lane);
      v8f sc[8];
#pragma unroll
      for (int nt = 0; nt < 8; ++nt) {  // A = K token-tiles
        v16h ak = frag_ld_k16(k16p, nt * 16, 128, hh * 16, lane);
        v8f z = {0.f, 0.f, 0.f, 0.f, 0.f, 0.f, 0.f, 0.f};
        sc[nt] = wmma_f16(ak, bq, z);   // S^T[token][query]
      }
      // softmax over tokens: 64 values in-lane + one cross-half shuffle
      float mx = -1e30f;
#pragma unroll
      for (int nt = 0; nt < 8; ++nt)
#pragma unroll
        for (int v = 0; v < 8; ++v) mx = fmaxf(mx, sc[nt][v] * scale);
      mx = fmaxf(mx, __shfl_xor(mx, 16, 32));
      float sum = 0.f;
#pragma unroll
      for (int nt = 0; nt < 8; ++nt)
#pragma unroll
        for (int v = 0; v < 8; ++v) {
          float e = __expf(sc[nt][v] * scale - mx);
          sc[nt][v] = e;
          sum += e;
        }
      sum += __shfl_xor(sum, 16, 32);
      float inv = 1.0f / sum;
      // D-layout of S^T == A-layout of P@V: build A-frags register->register
      v16h pa[4];
#pragma unroll
      for (int kb = 0; kb < 4; ++kb)
#pragma unroll
        for (int e = 0; e < 8; ++e) {
          pa[kb][e]     = (half_t)(sc[2 * kb][e] * inv);
          pa[kb][e + 8] = (half_t)(sc[2 * kb + 1][e] * inv);
        }
      v8f oacc = {0.f, 0.f, 0.f, 0.f, 0.f, 0.f, 0.f, 0.f};
#pragma unroll
      for (int kb = 0; kb < 4; ++kb) {
        v16h bv = frag_ld(vT16, hh * 16, 128, kb * 32, lane);
        oacc = wmma_f16(pa[kb], bv, oacc);
      }
#pragma unroll
      for (int v = 0; v < 8; ++v)
        o16[(mt * 16 + loff + v) * DM + hh * 16 + ncol] = (half_t)oacc[v];
    }
    __syncthreads();

    // ---- output projection + residual ----
    {
      const half_t* Wo = wout + l * 64 * 64;
      v16h a0 = frag_ld(o16, mt * 16, DM, 0, lane);
      v16h a1 = frag_ld(o16, mt * 16, DM, 32, lane);
      for (int nt = 0; nt < 4; ++nt) {
        v8f acc = {0.f, 0.f, 0.f, 0.f, 0.f, 0.f, 0.f, 0.f};
        v16h b0 = frag_ld(Wo, nt * 16, DM, 0, lane);
        v16h b1 = frag_ld(Wo, nt * 16, DM, 32, lane);
        acc = wmma_f16(a0, b0, acc);
        acc = wmma_f16(a1, b1, acc);
        int col = nt * 16 + ncol;
        float bb = bos[col];
#pragma unroll
        for (int v = 0; v < 8; ++v)
          hbuf[(mt * 16 + loff + v) * DM + col] += acc[v] + bb;
      }
    }
    __syncthreads();

    // ---- LayerNorm 1 (vectorized f16 writes) ----
    if (tid < KNB) {
      int r = tid;
      float m = 0.f;
      for (int d = 0; d < DM; ++d) m += hbuf[r * DM + d];
      m *= (1.0f / DM);
      float var = 0.f;
      for (int d = 0; d < DM; ++d) { float t = hbuf[r * DM + d] - m; var += t * t; }
      float inv = rsqrtf(var * (1.0f / DM) + EPSf);
      for (int db = 0; db < DM; db += 8) {
        v8h hv;
#pragma unroll
        for (int u = 0; u < 8; ++u) {
          int d = db + u;
          float y = (hbuf[r * DM + d] - m) * inv * ln1_s[l * 64 + d] + ln1_b[l * 64 + d];
          hbuf[r * DM + d] = y;
          hv[u] = (half_t)y;
        }
        *(v8h*)(h16 + r * DM + db) = hv;
      }
    }
    __syncthreads();

    // ---- FFN: ff1 computed transposed; relu + ff2 A-frags stay in registers ----
    {
      const half_t* W1 = wff1 + l * 128 * 64;
      const half_t* W2 = wff2 + l * 64 * 128;
      // B = this wave's h rows (lane = query col of ff1^T)
      v16h b0 = frag_ld(h16, mt * 16, DM, 0, lane);
      v16h b1 = frag_ld(h16, mt * 16, DM, 32, lane);
      v8f f[8];
#pragma unroll
      for (int nt = 0; nt < 8; ++nt) {     // A = W1 row-tiles (ff dim)
        v16h a0 = frag_ld(W1, nt * 16, DM, 0, lane);
        v16h a1 = frag_ld(W1, nt * 16, DM, 32, lane);
        v8f z = {0.f, 0.f, 0.f, 0.f, 0.f, 0.f, 0.f, 0.f};
        z = wmma_f16(a0, b0, z);
        f[nt] = wmma_f16(a1, b1, z);       // ff1^T[ffdim][query]
      }
      // bias (indexed by ff dim = D-row) + relu, packed into ff2 A-frags
      v16h pf[4];
#pragma unroll
      for (int kb = 0; kb < 4; ++kb)
#pragma unroll
        for (int e = 0; e < 8; ++e) {
          float x0 = f[2 * kb][e]     + bf1s[(2 * kb) * 16 + loff + e];
          float x1 = f[2 * kb + 1][e] + bf1s[(2 * kb + 1) * 16 + loff + e];
          pf[kb][e]     = (half_t)fmaxf(x0, 0.0f);
          pf[kb][e + 8] = (half_t)fmaxf(x1, 0.0f);
        }
      for (int nt = 0; nt < 4; ++nt) {     // ff2: out[m][d] + residual
        v8f acc = {0.f, 0.f, 0.f, 0.f, 0.f, 0.f, 0.f, 0.f};
#pragma unroll
        for (int kb = 0; kb < 4; ++kb) {
          v16h b = frag_ld(W2, nt * 16, 128, kb * 32, lane);
          acc = wmma_f16(pf[kb], b, acc);
        }
        int col = nt * 16 + ncol;
        float bb = bf2s[col];
#pragma unroll
        for (int v = 0; v < 8; ++v)
          hbuf[(mt * 16 + loff + v) * DM + col] += acc[v] + bb;
      }
    }
    __syncthreads();

    // ---- LayerNorm 2 ----
    if (tid < KNB) {
      int r = tid;
      float m = 0.f;
      for (int d = 0; d < DM; ++d) m += hbuf[r * DM + d];
      m *= (1.0f / DM);
      float var = 0.f;
      for (int d = 0; d < DM; ++d) { float t = hbuf[r * DM + d] - m; var += t * t; }
      float inv = rsqrtf(var * (1.0f / DM) + EPSf);
      for (int db = 0; db < DM; db += 8) {
        v8h hv;
#pragma unroll
        for (int u = 0; u < 8; ++u) {
          int d = db + u;
          float y = (hbuf[r * DM + d] - m) * inv * ln2_s[l * 64 + d] + ln2_b[l * 64 + d];
          hbuf[r * DM + d] = y;
          hv[u] = (half_t)y;
        }
        *(v8h*)(h16 + r * DM + db) = hv;
      }
    }
    __syncthreads();
  }

  // ---- pooled head ----
  if (tid < DM) {
    float s = 0.f;
    for (int r = 0; r < KNB; ++r) s += hbuf[r * DM + tid];
    gbuf[tid] = s * (1.0f / KNB);
  }
  __syncthreads();
  float hm = 0.f, hvar = 0.f;
  if (tid < DM) {
    for (int d = 0; d < DM; ++d) hm += gbuf[d];
    hm *= (1.0f / DM);
    for (int d = 0; d < DM; ++d) { float t = gbuf[d] - hm; hvar += t * t; }
    hvar *= (1.0f / DM);
  }
  __syncthreads();
  if (tid < DM)
    gbuf[tid] = (gbuf[tid] - hm) * rsqrtf(hvar + EPSf) * head_ln_s[tid] + head_ln_b[tid];
  __syncthreads();
  if (tid < DM) {
    float acc = head1_b[tid];
    for (int d = 0; d < DM; ++d) acc += gbuf[d] * head1_w[tid * DM + d];
    ybuf[tid] = 0.5f * acc * (1.0f + erff(acc * 0.70710678118654752f));  // exact gelu
  }
  __syncthreads();
  if (tid < 3) {
    float acc = head2_b[tid];
    for (int d = 0; d < DM; ++d) acc += ybuf[d] * head2_w[tid * DM + d];
    out[blk * 3 + tid] = acc;
  }
}

extern "C" void kernel_launch(void* const* d_in, const int* in_sizes, int n_in,
                              void* d_out, int out_size, void* d_ws, size_t ws_size,
                              hipStream_t stream) {
  (void)in_sizes; (void)n_in; (void)out_size; (void)ws_size;
  const int*   q_lin_idx = (const int*)d_in[0];
  const int*   offs      = (const int*)d_in[1];
  const float* coords    = (const float*)d_in[2];
  const float* vals      = (const float*)d_in[3];
  const float* log_gam   = (const float*)d_in[4];
  const float* proj_w    = (const float*)d_in[5];
  const float* proj_b    = (const float*)d_in[6];
  const float* qkv_w     = (const float*)d_in[7];
  const float* qkv_b     = (const float*)d_in[8];
  const float* out_w     = (const float*)d_in[9];
  const float* out_b     = (const float*)d_in[10];
  const float* ln1_s     = (const float*)d_in[11];
  const float* ln1_b     = (const float*)d_in[12];
  const float* ff1_w     = (const float*)d_in[13];
  const float* ff1_b     = (const float*)d_in[14];
  const float* ff2_w     = (const float*)d_in[15];
  const float* ff2_b     = (const float*)d_in[16];
  const float* ln2_s     = (const float*)d_in[17];
  const float* ln2_b     = (const float*)d_in[18];
  const float* hln_s     = (const float*)d_in[19];
  const float* hln_b     = (const float*)d_in[20];
  const float* head1_w   = (const float*)d_in[21];
  const float* head1_b   = (const float*)d_in[22];
  const float* head2_w   = (const float*)d_in[23];
  const float* head2_b   = (const float*)d_in[24];
  half_t* w16 = (half_t*)d_ws;
  float*  outp = (float*)d_out;

  ff_convert_weights<<<W16_TOT / 256, 256, 0, stream>>>(qkv_w, out_w, ff1_w, ff2_w, w16);
  ff_main<<<2048, 256, 0, stream>>>(q_lin_idx, offs, coords, vals, log_gam,
                                    proj_w, proj_b, qkv_b, out_b,
                                    ln1_s, ln1_b, ff1_b, ff2_b, ln2_s, ln2_b,
                                    hln_s, hln_b, head1_w, head1_b, head2_w, head2_b,
                                    w16, outp);
}
